// Block_57861799412251
// MI455X (gfx1250) — matrix operand order repaired and checked by reading, not compiled, not run on hardware
//
#include <hip/hip_runtime.h>
#include <stdint.h>

#define B_   32
#define CIN  64
#define CEXP 192
#define COUT 64
#define H_   128
#define W_   128
#define HW   (H_*W_)
#define EPS_ 1e-5f

// LDS K-strides padded for bank-conflict-free WMMA fragment loads:
// byte stride must stay 16B-aligned (for ds_load_b128) and have gcd(stride/4, 64)
// small.  72*2=144B (bank step 36), 200*2=400B (bank step 100).
#define XS_STR 72
#define XP_STR 200

typedef __attribute__((ext_vector_type(16))) __bf16 v16bf;
typedef __attribute__((ext_vector_type(8)))  float  v8f;

union AB16 { unsigned u[8]; v16bf v; };

__device__ __forceinline__ unsigned short f2bf(float f) {
    unsigned u = __float_as_uint(f);
    u += 0x7FFFu + ((u >> 16) & 1u);          // round-to-nearest-even
    return (unsigned short)(u >> 16);
}
__device__ __forceinline__ float bf2f(unsigned short h) {
    return __uint_as_float(((unsigned)h) << 16);
}

// ---------------------------------------------------------------------------
// One-shot: convert conv weights f32 -> bf16 in workspace.
// ---------------------------------------------------------------------------
__global__ void k_cvtw(const float* __restrict__ w, unsigned short* __restrict__ o, int n)
{
    int i = blockIdx.x * blockDim.x + threadIdx.x;
    if (i < n) o[i] = f2bf(w[i]);
}

// ---------------------------------------------------------------------------
// Kernel 1: expand 1x1 conv (GEMM M=128 tile, N=192, K=64) via bf16 WMMA,
// raw output -> t1 (bf16), fused per-(b,c) sum / sumsq stats.
// ---------------------------------------------------------------------------
__global__ __launch_bounds__(256) void k_expand(
    const float* __restrict__ x, const unsigned short* __restrict__ wbf,
    unsigned short* __restrict__ t1, float* __restrict__ s1, float* __restrict__ q1)
{
    __shared__ __align__(16) unsigned short xs [128*XS_STR];  // [m][k] bf16 (padded)
    __shared__ __align__(16) unsigned short wsh[192*XS_STR];  // [n][k] bf16 (padded)
    __shared__ float red[2*CEXP];

    const int tid = threadIdx.x;
    const int b   = blockIdx.y;
    const int m0  = blockIdx.x * 128;

    for (int i = tid; i < CEXP*CIN; i += 256) {
        int n = i >> 6, k = i & 63;
        wsh[n*XS_STR + k] = wbf[i];
    }
    const float* xb = x + (size_t)b * CIN * HW + m0;
    for (int i = tid; i < 128*64; i += 256) {
        int k = i >> 7, m = i & 127;
        xs[m*XS_STR + k] = f2bf(xb[(size_t)k * HW + m]);
    }
    for (int i = tid; i < 2*CEXP; i += 256) red[i] = 0.f;
    __syncthreads();

    const int wave = tid >> 5, lane = tid & 31;
    const int mw = wave * 16, lr = lane & 15, g = lane >> 4;

    v8f acc[12];
    const v8f z8 = {0.f,0.f,0.f,0.f,0.f,0.f,0.f,0.f};
    #pragma unroll
    for (int nt = 0; nt < 12; ++nt) acc[nt] = z8;

    #pragma unroll
    for (int kb = 0; kb < 2; ++kb) {
        AB16 A;
        const int mrow = mw + lr;
        #pragma unroll
        for (int j = 0; j < 8; ++j) {   // ISA 16-bit A 16x32 layout
            int koff = kb*32 + ((j >> 2) << 4) + (g << 3) + ((j & 3) << 1);
            A.u[j] = *(const unsigned*)&xs[mrow*XS_STR + koff];
        }
        #pragma unroll
        for (int nt = 0; nt < 12; ++nt) {
            AB16 Bm;
            const int n = nt*16 + lr;
            #pragma unroll
            for (int j = 0; j < 8; ++j) {  // ISA 16-bit B 32x16 layout
                int kk = kb*32 + ((j >> 2) << 4) + (g << 3) + ((j & 3) << 1);
                Bm.u[j] = *(const unsigned*)&wsh[n*XS_STR + kk];
            }
            acc[nt] = __builtin_amdgcn_wmma_f32_16x16x32_bf16(
                false, A.v, false, Bm.v, (short)0, acc[nt], false, false);
        }
    }

    #pragma unroll
    for (int nt = 0; nt < 12; ++nt) {
        const int n = nt*16 + lr;
        float s = 0.f, q = 0.f;
        unsigned pk[4];
        #pragma unroll
        for (int r = 0; r < 8; r += 2) {
            float v0 = acc[nt][r], v1 = acc[nt][r+1];
            s += v0 + v1; q += v0*v0 + v1*v1;
            pk[r >> 1] = (unsigned)f2bf(v0) | ((unsigned)f2bf(v1) << 16);
        }
        size_t off = ((size_t)b*CEXP + n)*HW + m0 + mw + g*8;
        *(uint4*)(t1 + off) = make_uint4(pk[0], pk[1], pk[2], pk[3]);
        s += __shfl_xor(s, 16);
        q += __shfl_xor(q, 16);
        if (g == 0) { atomicAdd(&red[n], s); atomicAdd(&red[CEXP + n], q); }
    }
    __syncthreads();
    if (tid < CEXP) {
        atomicAdd(&s1[b*CEXP + tid], red[tid]);
        atomicAdd(&q1[b*CEXP + tid], red[CEXP + tid]);
    }
}

// ---------------------------------------------------------------------------
// stats -> per-(b,c) scale/shift:  a = g/sqrt(var+eps), d = b - a*mean
// ---------------------------------------------------------------------------
__global__ void k_affine(const float* __restrict__ s, const float* __restrict__ q,
                         const float* __restrict__ gamma, const float* __restrict__ beta,
                         const int* __restrict__ ids, float* __restrict__ a,
                         float* __restrict__ d, int C)
{
    int i = blockIdx.x * blockDim.x + threadIdx.x;
    if (i >= B_ * C) return;
    int b = i / C, c = i - b * C;
    int dev = ids[b];
    float sv = s[i];
    float mean = sv * (1.f / HW);
    float var  = (q[i] - sv * mean) * (1.f / (HW - 1));   // unbiased (ddof=1)
    float ai = gamma[dev*C + c] * rsqrtf(var + EPS_);
    a[i] = ai;
    d[i] = beta[dev*C + c] - ai * mean;
}

// ---------------------------------------------------------------------------
// Kernel 3: normalize+ReLU t1 on the fly, depthwise 3x3, raw -> t2 (bf16),
// fused stats. One block per (b, c, 16-row stripe).
// ---------------------------------------------------------------------------
__global__ __launch_bounds__(256) void k_dw(
    const unsigned short* __restrict__ t1, const float* __restrict__ wdw,
    const float* __restrict__ a1, const float* __restrict__ d1,
    unsigned short* __restrict__ t2, float* __restrict__ s2, float* __restrict__ q2)
{
    __shared__ float ns[18*130];
    __shared__ float red2[2];
    const int tid = threadIdx.x;
    const int b = blockIdx.z, c = blockIdx.y, r0 = blockIdx.x * 16;
    const float av = a1[b*CEXP + c], dv = d1[b*CEXP + c];
    const size_t plane = ((size_t)b*CEXP + c) * HW;

    if (tid < 2) red2[tid] = 0.f;
    for (int i = tid; i < 18*130; i += 256) {
        int rr = i / 130, cc = i - rr*130;
        int row = r0 + rr - 1, col = cc - 1;
        float v = 0.f;
        if (row >= 0 && row < H_ && col >= 0 && col < W_) {
            float raw = bf2f(t1[plane + row*W_ + col]);
            v = fmaxf(av * raw + dv, 0.f);
        }
        ns[i] = v;
    }
    float wv[9];
    #pragma unroll
    for (int i = 0; i < 9; ++i) wv[i] = wdw[c*9 + i];
    __syncthreads();

    float s = 0.f, q = 0.f;
    #pragma unroll
    for (int k2 = 0; k2 < 8; ++k2) {
        int px = tid + k2*256;
        int r = px >> 7, col = px & 127;
        float acc = 0.f;
        #pragma unroll
        for (int dy = 0; dy < 3; ++dy)
            #pragma unroll
            for (int dx = 0; dx < 3; ++dx)
                acc = fmaf(wv[dy*3+dx], ns[(r+dy)*130 + col + dx], acc);
        t2[plane + (size_t)(r0 + r)*W_ + col] = f2bf(acc);
        s += acc; q += acc*acc;
    }
    #pragma unroll
    for (int m = 16; m >= 1; m >>= 1) { s += __shfl_xor(s, m); q += __shfl_xor(q, m); }
    if ((tid & 31) == 0) { atomicAdd(&red2[0], s); atomicAdd(&red2[1], q); }
    __syncthreads();
    if (tid == 0) atomicAdd(&s2[b*CEXP + c], red2[0]);
    if (tid == 1) atomicAdd(&q2[b*CEXP + c], red2[1]);
}

// ---------------------------------------------------------------------------
// Kernel 5: proj 1x1 conv (GEMM M=64 tile, N=64, K=192) via bf16 WMMA.
// Normalize+ReLU t2 while filling LDS. Raw out -> t3 (bf16), fused stats.
// ---------------------------------------------------------------------------
__global__ __launch_bounds__(128) void k_proj(
    const unsigned short* __restrict__ t2, const unsigned short* __restrict__ wbf,
    const float* __restrict__ a2, const float* __restrict__ d2,
    unsigned short* __restrict__ t3, float* __restrict__ s3, float* __restrict__ q3)
{
    __shared__ __align__(16) unsigned short xsp[64*XP_STR];   // [m][k] (padded)
    __shared__ __align__(16) unsigned short wsp[COUT*XP_STR]; // [n][k] (padded)
    __shared__ float as[CEXP], ds[CEXP];
    __shared__ float red[2*COUT];

    const int tid = threadIdx.x;
    const int b  = blockIdx.y;
    const int m0 = blockIdx.x * 64;

    for (int i = tid; i < CEXP; i += 128) { as[i] = a2[b*CEXP + i]; ds[i] = d2[b*CEXP + i]; }
    for (int i = tid; i < 2*COUT; i += 128) red[i] = 0.f;
    for (int i = tid; i < COUT*CEXP; i += 128) {
        int n = i / CEXP, k = i - n*CEXP;
        wsp[n*XP_STR + k] = wbf[i];
    }
    __syncthreads();   // as/ds ready before normalized fill

    const unsigned short* t2b = t2 + (size_t)b*CEXP*HW + m0;
    for (int i = tid; i < 64*CEXP; i += 128) {
        int k = i >> 6, m = i & 63;
        float raw = bf2f(t2b[(size_t)k*HW + m]);
        xsp[m*XP_STR + k] = f2bf(fmaxf(as[k]*raw + ds[k], 0.f));
    }
    __syncthreads();

    const int wave = tid >> 5, lane = tid & 31;
    const int mw = wave * 16, lr = lane & 15, g = lane >> 4;
    v8f acc[4];
    const v8f z8 = {0.f,0.f,0.f,0.f,0.f,0.f,0.f,0.f};
    #pragma unroll
    for (int nt = 0; nt < 4; ++nt) acc[nt] = z8;

    #pragma unroll
    for (int kb = 0; kb < 6; ++kb) {
        AB16 A;
        #pragma unroll
        for (int j = 0; j < 8; ++j) {
            int koff = kb*32 + ((j >> 2) << 4) + (g << 3) + ((j & 3) << 1);
            A.u[j] = *(const unsigned*)&xsp[(mw + lr)*XP_STR + koff];
        }
        #pragma unroll
        for (int nt = 0; nt < 4; ++nt) {
            AB16 Bm;
            const int n = nt*16 + lr;
            #pragma unroll
            for (int j = 0; j < 8; ++j)
                Bm.u[j] = *(const unsigned*)&wsp[n*XP_STR + kb*32 + ((j >> 2) << 4) + (g << 3) + ((j & 3) << 1)];
            acc[nt] = __builtin_amdgcn_wmma_f32_16x16x32_bf16(
                false, A.v, false, Bm.v, (short)0, acc[nt], false, false);
        }
    }

    #pragma unroll
    for (int nt = 0; nt < 4; ++nt) {
        const int n = nt*16 + lr;
        float s = 0.f, q = 0.f;
        unsigned pk[4];
        #pragma unroll
        for (int r = 0; r < 8; r += 2) {
            float v0 = acc[nt][r], v1 = acc[nt][r+1];
            s += v0 + v1; q += v0*v0 + v1*v1;
            pk[r >> 1] = (unsigned)f2bf(v0) | ((unsigned)f2bf(v1) << 16);
        }
        size_t off = ((size_t)b*COUT + n)*HW + m0 + mw + g*8;
        *(uint4*)(t3 + off) = make_uint4(pk[0], pk[1], pk[2], pk[3]);
        s += __shfl_xor(s, 16);
        q += __shfl_xor(q, 16);
        if (g == 0) { atomicAdd(&red[n], s); atomicAdd(&red[COUT + n], q); }
    }
    __syncthreads();
    if (tid < COUT)        atomicAdd(&s3[b*COUT + tid],        red[tid]);
    else if (tid < 2*COUT) atomicAdd(&q3[b*COUT + tid - COUT], red[tid]);
}

// ---------------------------------------------------------------------------
// Kernel 7: out = relu(a3*t3 + d3 + x), float4-vectorized.
// ---------------------------------------------------------------------------
__global__ __launch_bounds__(256) void k_final(
    const unsigned short* __restrict__ t3, const float* __restrict__ x,
    const float* __restrict__ a3, const float* __restrict__ d3,
    float* __restrict__ out)
{
    int gid = blockIdx.x * 256 + threadIdx.x;
    int i4 = gid * 4;
    int bc = i4 >> 14;              // / HW
    float av = a3[bc], dv = d3[bc];
    uint2 tp  = *(const uint2*)(t3 + i4);
    float4 xv = *(const float4*)(x + i4);
    float4 o;
    o.x = fmaxf(av * bf2f((unsigned short)(tp.x & 0xFFFFu)) + dv + xv.x, 0.f);
    o.y = fmaxf(av * bf2f((unsigned short)(tp.x >> 16))     + dv + xv.y, 0.f);
    o.z = fmaxf(av * bf2f((unsigned short)(tp.y & 0xFFFFu)) + dv + xv.z, 0.f);
    o.w = fmaxf(av * bf2f((unsigned short)(tp.y >> 16))     + dv + xv.w, 0.f);
    *(float4*)(out + i4) = o;
}

// ---------------------------------------------------------------------------
extern "C" void kernel_launch(void* const* d_in, const int* in_sizes, int n_in,
                              void* d_out, int out_size, void* d_ws, size_t ws_size,
                              hipStream_t stream)
{
    const float* x      = (const float*)d_in[0];
    const int*   ids    = (const int*)d_in[1];
    const float* w_exp  = (const float*)d_in[2];
    const float* g_exp  = (const float*)d_in[3];
    const float* b_exp  = (const float*)d_in[4];
    const float* w_dw   = (const float*)d_in[5];
    const float* g_dw   = (const float*)d_in[6];
    const float* b_dw   = (const float*)d_in[7];
    const float* w_proj = (const float*)d_in[8];
    const float* g_proj = (const float*)d_in[9];
    const float* b_proj = (const float*)d_in[10];
    float* out = (float*)d_out;

    char* ws = (char*)d_ws;
    const size_t T1 = (size_t)B_ * CEXP * HW * 2;        // 192 MiB bf16
    unsigned short* t1 = (unsigned short*)ws;
    unsigned short* t2 = (unsigned short*)(ws + T1);
    unsigned short* t3 = t1;                             // t1 dead after k_dw
    char* sb = ws + 2*T1;
    float* s1 = (float*)(sb);
    float* q1 = (float*)(sb +  24576);
    float* s2 = (float*)(sb +  49152);
    float* q2 = (float*)(sb +  73728);
    float* s3 = (float*)(sb +  98304);
    float* q3 = (float*)(sb + 106496);
    float* a1 = (float*)(sb + 114688);
    float* d1 = (float*)(sb + 139264);
    float* a2 = (float*)(sb + 163840);
    float* d2 = (float*)(sb + 188416);
    float* a3 = (float*)(sb + 212992);
    float* d3 = (float*)(sb + 221184);
    unsigned short* wexp_bf  = (unsigned short*)(sb + 229376);   // 24576 B
    unsigned short* wproj_bf = (unsigned short*)(sb + 253952);   // 24576 B

    hipMemsetAsync(sb, 0, 114688, stream);   // zero stats accumulators (capture-safe)

    k_cvtw<<<(CEXP*CIN  + 255)/256, 256, 0, stream>>>(w_exp,  wexp_bf,  CEXP*CIN);
    k_cvtw<<<(COUT*CEXP + 255)/256, 256, 0, stream>>>(w_proj, wproj_bf, COUT*CEXP);

    k_expand<<<dim3(HW/128, B_), 256, 0, stream>>>(x, wexp_bf, t1, s1, q1);
    k_affine<<<(B_*CEXP + 255)/256, 256, 0, stream>>>(s1, q1, g_exp, b_exp, ids, a1, d1, CEXP);
    k_dw    <<<dim3(H_/16, CEXP, B_), 256, 0, stream>>>(t1, w_dw, a1, d1, t2, s2, q2);
    k_affine<<<(B_*CEXP + 255)/256, 256, 0, stream>>>(s2, q2, g_dw, b_dw, ids, a2, d2, CEXP);
    k_proj  <<<dim3(HW/64, B_), 128, 0, stream>>>(t2, wproj_bf, a2, d2, t3, s3, q3);
    k_affine<<<(B_*COUT + 255)/256, 256, 0, stream>>>(s3, q3, g_proj, b_proj, ids, a3, d3, COUT);
    k_final <<<(B_*COUT*HW)/1024, 256, 0, stream>>>(t3, x, a3, d3, out);
}
